// MetaGRUCell_55843164783147
// MI455X (gfx1250) — matrix operand order repaired and checked by reading, not compiled, not run on hardware
//
#include <hip/hip_runtime.h>
#include <cstddef>

#define NN  207
#define BB  128
#define TT  12
#define HH  64
#define PHH 64

typedef __attribute__((ext_vector_type(16))) __bf16 v16bf;
typedef __attribute__((ext_vector_type(8)))  __bf16 v8bf;
typedef __attribute__((ext_vector_type(8)))  float  v8f;
typedef __attribute__((ext_vector_type(4)))  float  v4f;

__device__ __forceinline__ unsigned short tobf(float f) {
  return __builtin_bit_cast(unsigned short, (__bf16)f);
}
// fast sigmoid: v_exp_f32 + v_rcp_f32 (no IEEE divide chain)
__device__ __forceinline__ float sigm(float x) {
  return __builtin_amdgcn_rcpf(1.0f + __expf(-x));
}
// tanh(x) = 2*sigmoid(2x) - 1, same fast path
__device__ __forceinline__ float fast_tanh(float x) {
  return __builtin_fmaf(2.0f, __builtin_amdgcn_rcpf(1.0f + __expf(-2.0f * x)), -1.0f);
}

__device__ __forceinline__ v16bf cat16(v8bf lo, v8bf hi) {
  return __builtin_shufflevector(lo, hi, 0,1,2,3,4,5,6,7,8,9,10,11,12,13,14,15);
}
__device__ __forceinline__ v16bf frag16(const unsigned short* p) {
  v8bf lo = *(const v8bf*)(p);
  v8bf hi = *(const v8bf*)(p + 8);
  return cat16(lo, hi);
}
__device__ __forceinline__ v8f wmma_bf(v16bf a, v16bf b, v8f c) {
  return __builtin_amdgcn_wmma_f32_16x16x32_bf16(false, a, false, b, (short)0, c, false, false);
}

// Map (K,n) row-major index of a weight matrix (n-fastest, 64 cols) to its
// B-fragment-swizzled LDS slot: tile (kstep=K/32, ntile=n/16) stored as 1KB,
// lane (=16*((K%32/8)&1) + n%16) holds 16 bf16 in table order (e = K%8 + 8*(K%32>=16)).
__device__ __forceinline__ int swzB(int idx) {
  int K = idx >> 6, n = idx & 63;
  int kst = K >> 5, kk = K & 31;
  int half = (kk >> 3) & 1, e = (kk & 7) | ((kk >> 4) << 3);
  int nt = n >> 4,  nl = n & 15;
  return (kst * 4 + nt) * 512 + (half * 16 + nl) * 16 + e;
}

__global__ __launch_bounds__(256, 1)
void metagru_fused_kernel(const float* __restrict__ feature,
                          const float* __restrict__ inputs,
                          const float* __restrict__ states,
                          const float* __restrict__ W1,    const float* __restrict__ b1,
                          const float* __restrict__ W2,    const float* __restrict__ b2,
                          const float* __restrict__ W3_zr, const float* __restrict__ b3_zr,
                          const float* __restrict__ W3_ih, const float* __restrict__ b3_ih,
                          const float* __restrict__ W3b,   const float* __restrict__ b3b,
                          float* __restrict__ out) {
  // LDS layout (ushort units): Wz[0..8191] Wr[8192..16383] Wi[16384..20479] Wh[20480..24575]
  __shared__ __align__(16) unsigned short lds_w[24576];          // 48 KB bf16 fragments
  __shared__ __align__(16) unsigned short lds_stage[8][2][1024]; // per-wave s / r*s staging, 32 KB
  __shared__ float lds_h1[128];
  __shared__ float lds_h2[16];
  __shared__ float lds_bias[4];

  const int n   = blockIdx.x;
  const int tid = threadIdx.x;

  // ---------------- hypernetwork: shared 2-layer sigmoid MLP ----------------
  if (tid < 128) {
    int k = tid >> 4, mm = tid & 15;
    float acc = b1[k * 16 + mm];
    #pragma unroll
    for (int d = 0; d < 32; ++d)
      acc += feature[n * 32 + d] * W1[(k * 32 + d) * 16 + mm];
    lds_h1[tid] = sigm(acc);
  }
  __syncthreads();
  if (tid < 16) {
    int k = tid >> 1, p = tid & 1;
    float acc = b2[k * 2 + p];
    #pragma unroll
    for (int mm = 0; mm < 16; ++mm)
      acc += lds_h1[k * 16 + mm] * W2[(k * 16 + mm) * 2 + p];
    lds_h2[tid] = sigm(acc);
  }
  __syncthreads();
  if (tid < 4) {
    lds_bias[tid] = b3b[tid]
                  + lds_h2[(4 + tid) * 2 + 0] * W3b[tid * 2 + 0]
                  + lds_h2[(4 + tid) * 2 + 1] * W3b[tid * 2 + 1];
  }
  // ------------- generate per-node GRU weights straight into LDS -----------
  #pragma unroll
  for (int g = 0; g < 2; ++g) {            // Wz (h2[0]), Wr (h2[1]) : 128x64
    float c0 = lds_h2[g * 2 + 0], c1 = lds_h2[g * 2 + 1];
    const float* wa = W3_zr + (size_t)(g * 2 + 0) * 8192;
    const float* wb = W3_zr + (size_t)(g * 2 + 1) * 8192;
    const float* bb = b3_zr + (size_t)g * 8192;
    unsigned short* dst = &lds_w[g * 8192];
    for (int idx = tid; idx < 8192; idx += 256)
      dst[swzB(idx)] = tobf(bb[idx] + c0 * wa[idx] + c1 * wb[idx]);
  }
  #pragma unroll
  for (int g = 0; g < 2; ++g) {            // Wi (h2[2]), Wh (h2[3]) : 64x64
    float c0 = lds_h2[(2 + g) * 2 + 0], c1 = lds_h2[(2 + g) * 2 + 1];
    const float* wa = W3_ih + (size_t)(g * 2 + 0) * 4096;
    const float* wb = W3_ih + (size_t)(g * 2 + 1) * 4096;
    const float* bb = b3_ih + (size_t)g * 4096;
    unsigned short* dst = &lds_w[16384 + g * 4096];
    for (int idx = tid; idx < 4096; idx += 256)
      dst[swzB(idx)] = tobf(bb[idx] + c0 * wa[idx] + c1 * wb[idx]);
  }

  // ---------------- per-wave GRU over one 16-row batch strip ---------------
  const int w = tid >> 5, lane = tid & 31;
  const int m = lane & 15, half = lane >> 4;
  unsigned short* sStage  = &lds_stage[w][0][0];
  unsigned short* rsStage = &lds_stage[w][1][0];

  // initial state in C/D register layout + bf16 A-fragment staging
  v8f sreg[4];
  #pragma unroll
  for (int nt = 0; nt < 4; ++nt) {
    #pragma unroll
    for (int i = 0; i < 8; ++i) {
      int b   = w * 16 + i + half * 8;
      int col = nt * 16 + m;
      float v = states[((size_t)n * BB + b) * HH + col];
      sreg[nt][i] = v;
      sStage[((col >> 3) * 16 + (i + half * 8)) * 8 + (col & 7)] = tobf(v);
    }
  }
  __syncthreads();   // weights + staging visible; no barriers needed afterwards

  const float bz = lds_bias[0], br = lds_bias[1], bih = lds_bias[2] + lds_bias[3];
  const float* xrowbase = inputs + ((size_t)n * BB + (w * 16 + m)) * TT * PHH;
  float* finalOut = out + (size_t)NN * BB * TT * HH;
  const v8f vzero = {0.f, 0.f, 0.f, 0.f, 0.f, 0.f, 0.f, 0.f};

  for (int t = 0; t < TT; ++t) {
    const float* xrow = xrowbase + t * PHH;
    v8f zacc[4], racc[4], iacc[4];
    #pragma unroll
    for (int q = 0; q < 4; ++q) { zacc[q] = vzero; racc[q] = vzero; iacc[q] = vzero; }

    // x contribution: cat K-steps 0,1 feed Wz, Wr, Wi
    #pragma unroll
    for (int ks = 0; ks < 2; ++ks) {
      int c0 = ks * 32 + half * 8;
      v4f f0 = *(const v4f*)(xrow + c0);
      v4f f1 = *(const v4f*)(xrow + c0 + 4);
      v4f f2 = *(const v4f*)(xrow + c0 + 16);
      v4f f3 = *(const v4f*)(xrow + c0 + 20);
      v16bf a;
      #pragma unroll
      for (int e = 0; e < 4; ++e) {
        a[e]      = (__bf16)f0[e];
        a[e + 4]  = (__bf16)f1[e];
        a[e + 8]  = (__bf16)f2[e];
        a[e + 12] = (__bf16)f3[e];
      }
      #pragma unroll
      for (int nt = 0; nt < 4; ++nt) {
        zacc[nt] = wmma_bf(a, frag16(&lds_w[        (ks * 4 + nt) * 512 + lane * 16]), zacc[nt]);
        racc[nt] = wmma_bf(a, frag16(&lds_w[ 8192 + (ks * 4 + nt) * 512 + lane * 16]), racc[nt]);
        iacc[nt] = wmma_bf(a, frag16(&lds_w[16384 + (ks * 4 + nt) * 512 + lane * 16]), iacc[nt]);
      }
    }
    // state contribution: cat K-steps 2,3 feed Wz, Wr
    #pragma unroll
    for (int ks = 0; ks < 2; ++ks) {
      int g0 = ks * 4 + half, g1 = g0 + 2;
      v16bf a = cat16(*(const v8bf*)(&sStage[(g0 * 16 + m) * 8]),
                      *(const v8bf*)(&sStage[(g1 * 16 + m) * 8]));
      #pragma unroll
      for (int nt = 0; nt < 4; ++nt) {
        zacc[nt] = wmma_bf(a, frag16(&lds_w[       ((ks + 2) * 4 + nt) * 512 + lane * 16]), zacc[nt]);
        racc[nt] = wmma_bf(a, frag16(&lds_w[8192 + ((ks + 2) * 4 + nt) * 512 + lane * 16]), racc[nt]);
      }
    }
    // gates: z, r in fp32, stage r*s as bf16 A-fragments
    #pragma unroll
    for (int nt = 0; nt < 4; ++nt) {
      #pragma unroll
      for (int i = 0; i < 8; ++i) {
        float z = sigm(zacc[nt][i] + bz);
        float r = sigm(racc[nt][i] + br);
        zacc[nt][i] = z;
        float rs = r * sreg[nt][i];
        int col = nt * 16 + m;
        rsStage[((col >> 3) * 16 + (i + half * 8)) * 8 + (col & 7)] = tobf(rs);
      }
    }
    // hidden contribution: (r*s) @ Wh accumulated into iacc
    #pragma unroll
    for (int ks = 0; ks < 2; ++ks) {
      int g0 = ks * 4 + half, g1 = g0 + 2;
      v16bf a = cat16(*(const v8bf*)(&rsStage[(g0 * 16 + m) * 8]),
                      *(const v8bf*)(&rsStage[(g1 * 16 + m) * 8]));
      #pragma unroll
      for (int nt = 0; nt < 4; ++nt)
        iacc[nt] = wmma_bf(a, frag16(&lds_w[20480 + (ks * 4 + nt) * 512 + lane * 16]), iacc[nt]);
    }
    // GRU update, output store, restage new state
    #pragma unroll
    for (int nt = 0; nt < 4; ++nt) {
      #pragma unroll
      for (int i = 0; i < 8; ++i) {
        float cand = fast_tanh(iacc[nt][i] + bih);
        float z = zacc[nt][i];
        float sn = z * sreg[nt][i] + (1.0f - z) * cand;
        sreg[nt][i] = sn;
        int b   = w * 16 + i + half * 8;
        int col = nt * 16 + m;
        out[(((size_t)n * BB + b) * TT + t) * HH + col] = sn;
        sStage[((col >> 3) * 16 + (i + half * 8)) * 8 + (col & 7)] = tobf(sn);
      }
    }
  }
  // final state
  #pragma unroll
  for (int nt = 0; nt < 4; ++nt) {
    #pragma unroll
    for (int i = 0; i < 8; ++i) {
      int b   = w * 16 + i + half * 8;
      int col = nt * 16 + m;
      finalOut[((size_t)n * BB + b) * HH + col] = sreg[nt][i];
    }
  }
}

extern "C" void kernel_launch(void* const* d_in, const int* in_sizes, int n_in,
                              void* d_out, int out_size, void* d_ws, size_t ws_size,
                              hipStream_t stream) {
  (void)in_sizes; (void)n_in; (void)out_size; (void)d_ws; (void)ws_size;
  const float* feature = (const float*)d_in[0];
  const float* inputs  = (const float*)d_in[1];
  const float* states  = (const float*)d_in[2];
  const float* W1      = (const float*)d_in[3];
  const float* b1      = (const float*)d_in[4];
  const float* W2      = (const float*)d_in[5];
  const float* b2      = (const float*)d_in[6];
  const float* W3_zr   = (const float*)d_in[7];
  const float* b3_zr   = (const float*)d_in[8];
  const float* W3_ih   = (const float*)d_in[9];
  const float* b3_ih   = (const float*)d_in[10];
  const float* W3b     = (const float*)d_in[11];
  const float* b3b     = (const float*)d_in[12];
  float* out = (float*)d_out;

  metagru_fused_kernel<<<dim3(NN), dim3(256), 0, stream>>>(
      feature, inputs, states, W1, b1, W2, b2,
      W3_zr, b3_zr, W3_ih, b3_ih, W3b, b3b, out);
}
